// RNN_custom_61151744360680
// MI455X (gfx1250) — matrix operand order repaired and checked by reading, not compiled, not run on hardware
//
#include <hip/hip_runtime.h>
#include <hip/hip_bf16.h>

// ---------------------------------------------------------------------------
// SNN forward (adaptive LIF, 4 layers, 1024 steps) for MI455X / gfx1250.
// Single persistent workgroup, WMMA f16 (f32 accum) for all big GEMMs.
// Weights pre-packed to B-fragment order, L2-resident, re-fetched per step
// (pointer laundering blocks LICM; explicit addrspace(1) casts keep the
// laundered accesses on the GLOBAL path instead of FLAT).
// ---------------------------------------------------------------------------

typedef __attribute__((ext_vector_type(16))) _Float16 v16h;
typedef __attribute__((ext_vector_type(8)))  _Float16 v8h;
typedef __attribute__((ext_vector_type(8)))  float    v8f;

// explicit global-address-space views (avoid FLAT after pointer laundering)
typedef __attribute__((address_space(1))) const v16h*     gv16h_cp;
typedef __attribute__((address_space(1))) float*          gf32_p;
typedef __attribute__((address_space(1))) const _Float16* gf16_cp;

#define BATCH 128
#define NH    256
#define SEQ   2048
#define STEPS 1024
#define NOUT  20
#define KT_N  8      // 256 / 32 (K per WMMA)
#define NT_N  16     // 256 / 16 N tiles (hidden layers)

// -------------------- workspace layout (bytes) --------------------
#define WP_ONE   131072u               // one packed 256x256 f16 matrix
#define WP_OFF   0u                    // Wh2h, Wh2d, Wd2d, Wd1 packed (4x128KB)
#define WO_OFF   (4u * WP_ONE)        // packed output weights (16KB)
#define ST_OFF   (WO_OFF + 16384u)    // mem1,b1,mem2,b2,mem3,b3 (6x128KB f32)
#define ST_ONE   32768u               // floats per state array (128*256)
#define AT_OFF   (ST_OFF + 6u * 131072u)   // 8x256 f32 alpha/rho tables
#define WS_NEED  (AT_OFF + 8192u)

// dynamic LDS: s1,s2,s3 (f16) + xt + Wi2h + combined biases + tau tables
#define SMEM_BYTES (3*BATCH*NH*2 + BATCH*4*4 + NH*4*4 + 4*NH*4 + 8*NH*4)

// ---------------------------------------------------------------------------
// Prep kernel 1: pack the four 256x256 weight matrices into WMMA B-fragment
// order (f16). Fragment layout (dense 16x16x32 f16 B, 32x16 K-major):
//   n = nt*16 + (lane&15), k = kt*32 + (j>=8 ? 16 : 0) + (lane>=16 ? 8 : 0) + (j&7), j = 0..15.
// packed[((kt*16+nt)*32 + lane)*16 + j] = W[n][k]  (B[k][n] = W^T[k][n])
// ---------------------------------------------------------------------------
__global__ void pack_weights(const float* __restrict__ W0, const float* __restrict__ W1,
                             const float* __restrict__ W2, const float* __restrict__ W3,
                             _Float16* __restrict__ P) {
  int id = blockIdx.x * blockDim.x + threadIdx.x;
  if (id >= 4 * 65536) return;
  int mi = id >> 16;
  int r  = id & 65535;
  const float* W = (mi == 0) ? W0 : (mi == 1) ? W1 : (mi == 2) ? W2 : W3;
  int j    = r & 15;
  int lane = (r >> 4) & 31;
  int nt   = (r >> 9) & 15;
  int kt   = r >> 13;
  int k = kt * 32 + ((j & 8) ? 16 : 0) + ((lane & 16) ? 8 : 0) + (j & 7);
  int n = nt * 16 + (lane & 15);
  P[id] = (_Float16)W[n * NH + k];
}

// Output weights [20][256] -> padded to 32 columns, 8 kt x 2 nt tiles.
__global__ void pack_out_weights(const float* __restrict__ Wo, _Float16* __restrict__ P) {
  int id = blockIdx.x * blockDim.x + threadIdx.x;
  if (id >= 8192) return;
  int j    = id & 15;
  int lane = (id >> 4) & 31;
  int nt   = (id >> 9) & 1;
  int kt   = id >> 10;
  int k = kt * 32 + ((j & 8) ? 16 : 0) + ((lane & 16) ? 8 : 0) + (j & 7);
  int n = nt * 16 + (lane & 15);
  P[id] = (n < NOUT) ? (_Float16)Wo[n * NH + k] : (_Float16)0.0f;
}

// Prep kernel 2: init mem=u, b=b_j0 for the 3 hidden layers; build
// alpha = exp(-1/tau_m) and rho = exp(-1/tau_adp) tables (8 x 256).
__global__ void init_state(const float* __restrict__ u1, const float* __restrict__ u2,
                           const float* __restrict__ ud, float* __restrict__ st,
                           const float* __restrict__ tmr1, const float* __restrict__ tar1,
                           const float* __restrict__ tmr2, const float* __restrict__ tar2,
                           const float* __restrict__ tmd1, const float* __restrict__ tad1,
                           const float* __restrict__ tmo,  const float* __restrict__ tao,
                           float* __restrict__ aT) {
  int id = blockIdx.x * blockDim.x + threadIdx.x;
  if (id < 3 * 32768) {
    int layer = id >> 15, e = id & 32767;
    const float* u = (layer == 0) ? u1 : (layer == 1) ? u2 : ud;
    st[layer * 65536 + e]         = u[e];   // mem
    st[layer * 65536 + 32768 + e] = 0.01f;  // b = b_j0
  } else if (id < 3 * 32768 + 2048) {
    int q = id - 3 * 32768;
    int a = q >> 8, n = q & 255;
    const float* tau = (a == 0) ? tmr1 : (a == 1) ? tar1 : (a == 2) ? tmr2 :
                       (a == 3) ? tar2 : (a == 4) ? tmd1 : (a == 5) ? tad1 :
                       (a == 6) ? tmo  : tao;
    float v = 0.0f;
    if (a < 6 || n < NOUT) v = expf(-1.0f / tau[n]);
    aT[q] = v;
  }
}

// ---------------------------------------------------------------------------
// Fragment loaders. A (spikes) from LDS row-major [128][256] f16 per the
// 16-bit A layout: lanes 0-15 row m, j0..7 -> K 0..7, j8..15 -> K 16..23;
// lanes 16-31 shift K by +8. One base VGPR; all tile/kt offsets are
// compile-time constants folded into the DS immediate (mt stride = 4096 elem).
// ---------------------------------------------------------------------------
__device__ __forceinline__ v16h load_a_off(const _Float16* base, int elemOff) {
  v8h lo = *(const v8h*)(base + elemOff);
  v8h hi = *(const v8h*)(base + elemOff + 16);
  return __builtin_shufflevector(lo, hi, 0, 1, 2, 3, 4, 5, 6, 7,
                                 8, 9, 10, 11, 12, 13, 14, 15);
}

// B fragment: 16 contiguous halves per lane, via explicit global addrspace.
__device__ __forceinline__ v16h load_b_g(const _Float16* wp, int elemOff) {
  gf16_cp g = (gf16_cp)(unsigned long long)wp;
  return *(gv16h_cp)(g + elemOff);
}

// Wave owns a fixed N tile (boff encodes nt+lane) and 4 consecutive M tiles:
// one global B fragment per kt feeds four WMMAs (A comes from LDS).
__device__ __forceinline__ void gemm_acc(v8f c[4], const _Float16* aBase,
                                         const _Float16* wp, int boff) {
#pragma unroll
  for (int kt = 0; kt < KT_N; ++kt) {
    v16h b = load_b_g(wp, boff + kt * (NT_N * 32 * 16));
#pragma unroll
    for (int i = 0; i < 4; ++i) {
      v16h a = load_a_off(aBase, i * (16 * NH) + kt * 32);
      c[i] = __builtin_amdgcn_wmma_f32_16x16x32_f16(false, a, false, b,
                                                    (short)0, c[i], false, false);
    }
  }
}

// Adaptive-LIF elementwise update for one hidden layer. C layout: VGPR r,
// lanes 0-15 -> M=r, lanes 16-31 -> M=r+8; column n = nt*16 + (lane&15)
// is wave-constant, so per-neuron alpha/rho/bias are hoisted.
__device__ __forceinline__ void update_layer(const v8f c[4], int li,
    _Float16* __restrict__ sbuf, float* memG, float* bG,
    const float* __restrict__ aTL, const float* __restrict__ hbL,
    const float* __restrict__ xtL, const float* __restrict__ wiL, bool addInput,
    int mtb, int nt, int col, int hiM) {
  const int   n  = nt * 16 + col;
  const float an = aTL[(2 * li)     * NH + n];
  const float rn = aTL[(2 * li + 1) * NH + n];
  const float hb = hbL[li * NH + n];
  float w0 = 0.f, w1 = 0.f, w2 = 0.f, w3 = 0.f;
  if (addInput) { w0 = wiL[n*4+0]; w1 = wiL[n*4+1]; w2 = wiL[n*4+2]; w3 = wiL[n*4+3]; }
  gf32_p mg = (gf32_p)(unsigned long long)memG;
  gf32_p bg = (gf32_p)(unsigned long long)bG;
#pragma unroll
  for (int i = 0; i < 4; ++i) {
    const int mt = mtb + i;
#pragma unroll
    for (int r = 0; r < 8; ++r) {
      const int m   = mt * 16 + r + hiM;
      const int idx = m * NH + n;
      float h = c[i][r] + hb;
      if (addInput)
        h += xtL[m*4+0]*w0 + xtL[m*4+1]*w1 + xtL[m*4+2]*w2 + xtL[m*4+3]*w3;
      float mem = mg[idx];
      float b   = bg[idx];
      const float spk = (float)sbuf[idx];
      b = rn * b + (1.0f - rn) * spk;                // rho*b + (1-rho)*spk
      const float Bth = 0.01f + 1.8f * b;            // b_j0 + beta*b
      mem = mem * an + (1.0f - an) * h - Bth * spk;  // R_m=1, dt=1
      const float sn = (mem - Bth > 0.0f) ? 1.0f : 0.0f;
      mg[idx]   = mem;
      bg[idx]   = b;
      sbuf[idx] = (_Float16)sn;
    }
  }
}

// ---------------------------------------------------------------------------
// Main persistent kernel: 1 workgroup, 32 waves, 1024 steps in-kernel.
// ---------------------------------------------------------------------------
__global__ __launch_bounds__(1024, 1) void snn_forward(
    const float* __restrict__ x,
    const float* __restrict__ Wi2h, const float* __restrict__ bi2h,
    const float* __restrict__ bh2h, const float* __restrict__ bh2d,
    const float* __restrict__ bd2d, const float* __restrict__ bd1v,
    const float* __restrict__ bd2o,
    const float* __restrict__ u1, const float* __restrict__ u2,
    const float* __restrict__ ud, const float* __restrict__ uo,
    const _Float16* __restrict__ Wp,   // packed Wh2h,Wh2d,Wd2d,Wd1
    const _Float16* __restrict__ WoP,  // packed output weights
    float* st,                          // 6 state arrays
    const float* __restrict__ aT,      // 8x256 alpha/rho tables
    float* out) {
  extern __shared__ char smem[];
  _Float16* s1  = (_Float16*)smem;
  _Float16* s2  = s1 + BATCH * NH;
  _Float16* s3  = s2 + BATCH * NH;
  float*    xtL = (float*)(s3 + BATCH * NH);
  float*    wiL = xtL + BATCH * 4;
  float*    hbL = wiL + NH * 4;
  float*    aTL = hbL + 4 * NH;

  const int tid  = threadIdx.x;
  const int w    = tid >> 5;
  const int lane = tid & 31;
  const int col  = lane & 15;
  const int hiX  = (lane & 16) ? 8 : 0;  // row offset (C) and K offset (A)

  float* mem1 = st;                float* b1 = st + 1 * ST_ONE;
  float* mem2 = st + 2 * ST_ONE;   float* b2 = st + 3 * ST_ONE;
  float* mem3 = st + 4 * ST_ONE;   float* b3 = st + 5 * ST_ONE;

  // stage small constants + initial spikes into LDS
  for (int i = tid; i < NH * 4; i += 1024) wiL[i] = Wi2h[i];
  for (int i = tid; i < 8 * NH; i += 1024) aTL[i] = aT[i];
  if (tid < NH) {
    hbL[0 * NH + tid] = bi2h[tid] + bh2h[tid];
    hbL[1 * NH + tid] = bh2d[tid] + bd2d[tid];
    hbL[2 * NH + tid] = bd1v[tid];
    hbL[3 * NH + tid] = (tid < NOUT) ? bd2o[tid] : 0.0f;
  }
  for (int i = tid; i < BATCH * NH; i += 1024) {
    s1[i] = (_Float16)u1[i];
    s2[i] = (_Float16)u2[i];
    s3[i] = (_Float16)ud[i];
  }

  // hidden-layer tile map: fixed N tile per wave, 4 M tiles
  const int nt  = w >> 1;         // 0..15
  const int mtb = (w & 1) * 4;    // 0 or 4

  // per-wave constant bases (1 VGPR each; all tile offsets are immediates)
  const int aoff = (mtb * 16 + col) * NH + hiX;
  const _Float16* a1B = s1 + aoff;
  const _Float16* a2B = s2 + aoff;
  const int boff = (nt * 32 + lane) * 16;            // hidden B fragment base

  // output-layer state in registers (waves 0..15, one 16x16 tile each)
  const int  omt = w & 7, ont = w >> 3;              // valid for w < 16
  const int  on  = ont * 16 + col;
  const bool ow  = (w < 16);
  const bool ovalid = ow && (on < NOUT);
  const _Float16* aoB   = s3 + (omt * 16 + col) * NH + hiX;
  const int       boffO = (ont * 32 + lane) * 16;
  float om[8], osp[8], bo[8], osum[8];
#pragma unroll
  for (int r = 0; r < 8; ++r) {
    const int   m = omt * 16 + r + hiX;
    const float v = ovalid ? uo[m * NOUT + on] : 0.0f;
    om[r] = v; osp[r] = v; osum[r] = v; bo[r] = 0.01f;
  }
  __syncthreads();

  const v8f z = {0.f, 0.f, 0.f, 0.f, 0.f, 0.f, 0.f, 0.f};

  // Launderable weight pointers: the empty asm below makes their values
  // opaque each iteration so LICM cannot hoist the (loop-invariant) weight
  // fragment loads out of the 1024-step loop and spill them to scratch.
  const _Float16* WpL  = Wp;
  const _Float16* WoPL = WoP;

#pragma unroll 1
  for (int t = 0; t < STEPS; ++t) {
    asm volatile("" : "+s"(WpL), "+s"(WoPL));

    // stage x window [128][4] (start = min(2t, SEQ-4)), scaled by 1/255
    if (tid < BATCH * 4) {
      const int m = tid >> 2, k = tid & 3;
      int start = 2 * t; if (start > SEQ - 4) start = SEQ - 4;
      xtL[tid] = x[m * SEQ + start + k] * (1.0f / 255.0f);
    }
    __syncthreads();

    v8f c[4];

    // ---- layer 1: h = s1 @ Wh2h^T (+ x@Wi2h^T + biases in update)
    c[0] = z; c[1] = z; c[2] = z; c[3] = z;
    gemm_acc(c, a1B, WpL + 0 * 65536, boff);
    __syncthreads();                               // all s1 reads done
    update_layer(c, 0, s1, mem1, b1, aTL, hbL, xtL, wiL, true, mtb, nt, col, hiX);
    __syncthreads();                               // new s1 visible

    // ---- layer 2: d = s1_new @ Wh2d^T + s2 @ Wd2d^T
    c[0] = z; c[1] = z; c[2] = z; c[3] = z;
    gemm_acc(c, a1B, WpL + 1 * 65536, boff);
    gemm_acc(c, a2B, WpL + 2 * 65536, boff);
    __syncthreads();
    update_layer(c, 1, s2, mem2, b2, aTL, hbL, xtL, wiL, false, mtb, nt, col, hiX);
    __syncthreads();

    // ---- layer 3: s2_new @ Wd1^T
    c[0] = z; c[1] = z; c[2] = z; c[3] = z;
    gemm_acc(c, a2B, WpL + 3 * 65536, boff);
    __syncthreads();
    update_layer(c, 2, s3, mem3, b3, aTL, hbL, xtL, wiL, false, mtb, nt, col, hiX);
    __syncthreads();

    // ---- output layer (waves 0..15; wave-uniform branch keeps EXEC full)
    if (ow) {
      v8f co = z;
#pragma unroll
      for (int kt = 0; kt < KT_N; ++kt) {
        v16h b = load_b_g(WoPL, boffO + kt * (2 * 32 * 16));
        v16h a = load_a_off(aoB, kt * 32);
        co = __builtin_amdgcn_wmma_f32_16x16x32_f16(false, a, false, b,
                                                    (short)0, co, false, false);
      }
      if (ovalid) {
        const float ao = aTL[6 * NH + on];
        const float ro = aTL[7 * NH + on];
        const float hb = hbL[3 * NH + on];
#pragma unroll
        for (int r = 0; r < 8; ++r) {
          const float oin = co[r] + hb;
          bo[r] = ro * bo[r] + (1.0f - ro) * osp[r];
          const float Bth = 0.01f + 1.8f * bo[r];
          om[r] = om[r] * ao + (1.0f - ao) * oin - Bth * osp[r];
          osp[r] = (om[r] - Bth > 0.0f) ? 1.0f : 0.0f;
          osum[r] += osp[r];
        }
      }
    }
    // next write to s3 is after 6 barriers in t+1, so the output GEMM's
    // s3 reads are safe; the xt re-stage touches only xtL.
  }

  if (ovalid) {
    gf32_p og = (gf32_p)(unsigned long long)out;
#pragma unroll
    for (int r = 0; r < 8; ++r) {
      const int m = omt * 16 + r + hiX;
      og[m * NOUT + on] = osum[r];
    }
  }
}

// ---------------------------------------------------------------------------
extern "C" void kernel_launch(void* const* d_in, const int* in_sizes, int n_in,
                              void* d_out, int out_size, void* d_ws, size_t ws_size,
                              hipStream_t stream) {
  (void)in_sizes; (void)n_in; (void)out_size; (void)ws_size;

  const float* x    = (const float*)d_in[0];
  const float* Wi2h = (const float*)d_in[1];
  const float* bi2h = (const float*)d_in[2];
  const float* Wh2h = (const float*)d_in[3];
  const float* bh2h = (const float*)d_in[4];
  const float* Wh2d = (const float*)d_in[5];
  const float* bh2d = (const float*)d_in[6];
  const float* Wd2d = (const float*)d_in[7];
  const float* bd2d = (const float*)d_in[8];
  const float* Wd1  = (const float*)d_in[9];
  const float* bd1v = (const float*)d_in[10];
  const float* Wd2o = (const float*)d_in[11];
  const float* bd2o = (const float*)d_in[12];
  const float* tar1 = (const float*)d_in[13];
  const float* tar2 = (const float*)d_in[14];
  const float* tad1 = (const float*)d_in[15];
  const float* tao  = (const float*)d_in[16];
  const float* tmr1 = (const float*)d_in[17];
  const float* tmr2 = (const float*)d_in[18];
  const float* tmd1 = (const float*)d_in[19];
  const float* tmo  = (const float*)d_in[20];
  const float* u1   = (const float*)d_in[21];
  const float* u2   = (const float*)d_in[22];
  const float* ud   = (const float*)d_in[23];
  const float* uo   = (const float*)d_in[24];

  char*     ws  = (char*)d_ws;
  _Float16* Wp  = (_Float16*)(ws + WP_OFF);
  _Float16* WoP = (_Float16*)(ws + WO_OFF);
  float*    st  = (float*)(ws + ST_OFF);
  float*    aT  = (float*)(ws + AT_OFF);

  pack_weights<<<1024, 256, 0, stream>>>(Wh2h, Wh2d, Wd2d, Wd1, Wp);
  pack_out_weights<<<32, 256, 0, stream>>>(Wd2o, WoP);
  init_state<<<(3 * 32768 + 2048 + 255) / 256, 256, 0, stream>>>(
      u1, u2, ud, st, tmr1, tar1, tmr2, tar2, tmd1, tad1, tmo, tao, aT);

  hipFuncSetAttribute((const void*)snn_forward,
                      hipFuncAttributeMaxDynamicSharedMemorySize, SMEM_BYTES);

  snn_forward<<<1, 1024, SMEM_BYTES, stream>>>(
      x, Wi2h, bi2h, bh2h, bh2d, bd2d, bd1v, bd2o,
      u1, u2, ud, uo, Wp, WoP, st, aT, (float*)d_out);
}
